// EdgeDegreeEmbedding_40742059770591
// MI455X (gfx1250) — compile-verified, matrix operand order checked by reading, not run on hardware
//
#include <hip/hip_runtime.h>
#include <hip/hip_bf16.h>

typedef __attribute__((ext_vector_type(16))) __bf16 v16bf;
typedef __attribute__((ext_vector_type(8)))  float  v8f;
typedef __attribute__((ext_vector_type(4)))  int    v4i;

#define E_EDGES   50000
#define NNODES    5000
#define D_DIST    512
#define D_IN      768
#define HID       128
#define M0_DIM    7
#define SPH       128
#define OUT3      (M0_DIM*SPH)   /* 896 */
#define LFULL     49
#define M_ALL     29
#define CUTOFF    12.0f
#define RESCALE   23.395238876342773f

__device__ __forceinline__ unsigned short f32_to_bf16u(float f) {
  union { float f; unsigned u; } v; v.f = f;
  unsigned u = v.u;
  u += 0x7FFFu + ((u >> 16) & 1u);   // round-to-nearest-even
  return (unsigned short)(u >> 16);
}
__device__ __forceinline__ __bf16 us2bf(unsigned short s) {
  return __builtin_bit_cast(__bf16, s);
}

// ---- CDNA5 async global->LDS copy (ASYNCcnt path), 16 bytes per lane ------
__device__ __forceinline__ void async_copy16(const float* g, float* lds) {
#if defined(__HIP_DEVICE_COMPILE__) && __has_builtin(__builtin_amdgcn_global_load_async_to_lds_b128)
  typedef __attribute__((address_space(1))) v4i gv4i_t;
  typedef __attribute__((address_space(3))) v4i lv4i_t;
  __builtin_amdgcn_global_load_async_to_lds_b128(
      (gv4i_t*)(uintptr_t)g,
      (lv4i_t*)(unsigned)(uintptr_t)lds,
      0, 0);
#elif defined(__HIP_DEVICE_COMPILE__)
  // GV mode: vdst = LDS byte offset (low 32 bits of generic LDS address),
  // vaddr = 64-bit global address.
  unsigned ldsoff = (unsigned)(uintptr_t)lds;
  asm volatile("global_load_async_to_lds_b128 %0, %1, off"
               :: "v"(ldsoff), "v"(g) : "memory");
#else
  *(float4*)lds = *(const float4*)g;
#endif
}
__device__ __forceinline__ void wait_async0() {
#if defined(__HIP_DEVICE_COMPILE__) && __has_builtin(__builtin_amdgcn_s_wait_asynccnt)
  __builtin_amdgcn_s_wait_asynccnt(0);
#elif defined(__HIP_DEVICE_COMPILE__)
  asm volatile("s_wait_asynccnt 0" ::: "memory");
#endif
}

// ---------------------------------------------------------------- weights
// Store bf16 weights transposed: WT[n][k] so B-fragments read contiguous K.
__global__ void convert_weights_kernel(const float* __restrict__ W1,
                                       const float* __restrict__ W2,
                                       const float* __restrict__ W3,
                                       unsigned short* __restrict__ W1T,
                                       unsigned short* __restrict__ W2T,
                                       unsigned short* __restrict__ W3T) {
  int stride = gridDim.x * blockDim.x;
  int g = blockIdx.x * blockDim.x + threadIdx.x;
  for (int i = g; i < HID * D_IN; i += stride) {          // W1: [768,128] -> [128][768]
    int c = i / D_IN, k = i - c * D_IN;
    W1T[i] = f32_to_bf16u(W1[(size_t)k * HID + c]);
  }
  for (int i = g; i < HID * HID; i += stride) {           // W2: [128,128] -> [128][128]
    int c = i / HID, k = i - c * HID;
    W2T[i] = f32_to_bf16u(W2[(size_t)k * HID + c]);
  }
  for (int i = g; i < OUT3 * HID; i += stride) {          // W3: [128,896] -> [896][128]
    int c = i / HID, k = i - c * HID;
    W3T[i] = f32_to_bf16u(W3[(size_t)k * OUT3 + c]);
  }
}

// ---------------------------------------------------------------- out = x
__global__ void copy_x_kernel(const float4* __restrict__ x, float4* __restrict__ out, int n4) {
  int stride = gridDim.x * blockDim.x;
  for (int i = blockIdx.x * blockDim.x + threadIdx.x; i < n4; i += stride)
    out[i] = x[i];
}

// -------- parallel LayerNorm + SiLU: 16 lanes per row, shfl_xor butterfly --
__device__ __forceinline__ void ln_silu_all(const float (*hf)[HID],
                                            unsigned short (*sH)[HID],
                                            const float* __restrict__ g,
                                            const float* __restrict__ bt,
                                            int tid) {
  int row = tid >> 4, sub = tid & 15;
  float v[8], s = 0.f;
  #pragma unroll
  for (int j = 0; j < 8; ++j) { v[j] = hf[row][sub + 16 * j]; s += v[j]; }
  #pragma unroll
  for (int m = 8; m >= 1; m >>= 1) s += __shfl_xor(s, m, 16);
  float mu = s * (1.0f / HID);
  float q = 0.f;
  #pragma unroll
  for (int j = 0; j < 8; ++j) { float d = v[j] - mu; q += d * d; }
  #pragma unroll
  for (int m = 8; m >= 1; m >>= 1) q += __shfl_xor(q, m, 16);
  float rs = rsqrtf(q * (1.0f / HID) + 1e-5f);
  #pragma unroll
  for (int j = 0; j < 8; ++j) {
    int c = sub + 16 * j;
    float t = (v[j] - mu) * rs * g[c] + bt[c];
    float si = t / (1.0f + __expf(-t));
    sH[row][c] = f32_to_bf16u(si);
  }
}

// ---------------------------------------------------------------- fused edge kernel
// 1 workgroup = 16 edges, 256 threads = 8 wave32.
__global__ __launch_bounds__(256) void fused_edge_kernel(
    const float* __restrict__ ede,  const float* __restrict__ semb,
    const float* __restrict__ temb, const float* __restrict__ edist,
    const int*   __restrict__ eidx, const float* __restrict__ wig,
    const unsigned short* __restrict__ W1T, const float* __restrict__ b1,
    const float* __restrict__ g1,  const float* __restrict__ bt1,
    const unsigned short* __restrict__ W2T, const float* __restrict__ b2,
    const float* __restrict__ g2,  const float* __restrict__ bt2,
    const unsigned short* __restrict__ W3T, const float* __restrict__ b3,
    float* __restrict__ out)
{
  __shared__ __align__(16) float          sXf[16 * D_IN];      // 49152 B async staging
  // union region: sA [16][768] bf16 (GEMM1 A), later sM0 [16][128][7] bf16
  __shared__ __align__(16) unsigned short sU[16 * OUT3];       // 28672 B
  __shared__ __align__(16) float          hf[16][HID];         //  8192 B
  __shared__ __align__(16) unsigned short sH[16][HID];         //  4096 B
  __shared__ float sScale[16];

  unsigned short* sA  = sU;   // [le][768]
  unsigned short* sM0 = sU;   // [le][c][m] stride 7

  const int tid  = threadIdx.x;
  const int w    = tid >> 5;
  const int lane = tid & 31;
  const int hi   = (lane >> 4) & 1;     // which 16-lane half
  const int lr   = lane & 15;
  const int e0   = blockIdx.x * 16;     // E_EDGES % 16 == 0 -> all blocks full

  // per-edge envelope scale (folded into m0: einsum is linear in m0)
  if (tid < 16) {
    int e = e0 + tid;
    float d  = edist[e] * (1.0f / CUTOFF);
    float d2 = d * d, d4 = d2 * d2, d5 = d4 * d, d6 = d5 * d, d7 = d6 * d;
    float env = 1.0f - 21.0f * d5 + 35.0f * d6 - 15.0f * d7;
    env = (d < 1.0f) ? env : 0.0f;
    sScale[tid] = env / RESCALE;
  }

  // ---- async gather concat(dist_emb, src_emb, tgt_emb) -> f32 LDS staging
  for (int i = tid; i < 16 * (D_IN / 4); i += 256) {    // 192 16-byte chunks/edge
    int le = i / (D_IN / 4), q = i - le * (D_IN / 4);
    int col = q << 2;
    int e = e0 + le;
    const float* src;
    if (col < D_DIST)            src = ede  + (size_t)e * D_DIST + col;
    else if (col < D_DIST + 128) src = semb + (size_t)eidx[e] * 128 + (col - D_DIST);
    else                         src = temb + (size_t)eidx[E_EDGES + e] * 128 + (col - D_DIST - 128);
    async_copy16(src, &sXf[le * D_IN + col]);
  }
  wait_async0();
  __syncthreads();

  // ---- convert staged f32 -> bf16 (packed pair stores)
  {
    unsigned* sA32 = (unsigned*)sA;
    for (int i = tid; i < 16 * D_IN / 2; i += 256) {
      float2 v = *(const float2*)&sXf[2 * i];
      sA32[i] = (unsigned)f32_to_bf16u(v.x) | ((unsigned)f32_to_bf16u(v.y) << 16);
    }
  }
  __syncthreads();

  // ---- GEMM1: [16x768]@[768x128], wave w owns N-tile w (cols 16w..16w+15)
  {
    v8f acc = {0.f, 0.f, 0.f, 0.f, 0.f, 0.f, 0.f, 0.f};
    const unsigned short* arow = sA + lr * D_IN;
    const unsigned short* bcol = W1T + (size_t)(w * 16 + lr) * D_IN;
    for (int k = 0; k < D_IN / 32; ++k) {
      int K0 = k * 32 + hi * 8;
      v16bf a, b;
      #pragma unroll
      for (int i = 0; i < 16; ++i) {
        int K = K0 + ((i >> 3) << 4) + (i & 7);   // 16-bit A layout K pattern
        a[i] = us2bf(arow[K]);
        b[i] = us2bf(bcol[K]);
      }
      acc = __builtin_amdgcn_wmma_f32_16x16x32_bf16(false, a, false, b, (short)0, acc, false, false);
    }
    int col = w * 16 + lr;
    #pragma unroll
    for (int r = 0; r < 8; ++r) hf[hi * 8 + r][col] = acc[r] + b1[col];
  }
  __syncthreads();
  ln_silu_all(hf, sH, g1, bt1, tid);
  __syncthreads();

  // ---- GEMM2: [16x128]@[128x128]
  {
    v8f acc = {0.f, 0.f, 0.f, 0.f, 0.f, 0.f, 0.f, 0.f};
    const unsigned short* arow = &sH[lr][0];
    const unsigned short* bcol = W2T + (size_t)(w * 16 + lr) * HID;
    for (int k = 0; k < HID / 32; ++k) {
      int K0 = k * 32 + hi * 8;
      v16bf a, b;
      #pragma unroll
      for (int i = 0; i < 16; ++i) {
        int K = K0 + ((i >> 3) << 4) + (i & 7);
        a[i] = us2bf(arow[K]);
        b[i] = us2bf(bcol[K]);
      }
      acc = __builtin_amdgcn_wmma_f32_16x16x32_bf16(false, a, false, b, (short)0, acc, false, false);
    }
    int col = w * 16 + lr;
    #pragma unroll
    for (int r = 0; r < 8; ++r) hf[hi * 8 + r][col] = acc[r] + b2[col];
  }
  __syncthreads();
  ln_silu_all(hf, sH, g2, bt2, tid);
  __syncthreads();

  // ---- GEMM3: [16x128]@[128x896]; wave w owns N-tiles {w, w+8, ..., w+48}
  {
    v8f acc[7];
    #pragma unroll
    for (int t = 0; t < 7; ++t) acc[t] = (v8f){0.f, 0.f, 0.f, 0.f, 0.f, 0.f, 0.f, 0.f};
    const unsigned short* arow = &sH[lr][0];
    for (int k = 0; k < HID / 32; ++k) {
      int K0 = k * 32 + hi * 8;
      v16bf a;
      #pragma unroll
      for (int i = 0; i < 16; ++i) a[i] = us2bf(arow[K0 + ((i >> 3) << 4) + (i & 7)]);
      #pragma unroll
      for (int t = 0; t < 7; ++t) {
        int col = (t * 8 + w) * 16 + lr;
        const unsigned short* bcol = W3T + (size_t)col * HID;
        v16bf b;
        #pragma unroll
        for (int i = 0; i < 16; ++i) b[i] = us2bf(bcol[K0 + ((i >> 3) << 4) + (i & 7)]);
        acc[t] = __builtin_amdgcn_wmma_f32_16x16x32_bf16(false, a, false, b, (short)0, acc[t], false, false);
      }
    }
    // bias + envelope scale, store m0 transposed bf16: sM0[le][c][m]
    #pragma unroll
    for (int t = 0; t < 7; ++t) {
      int col = (t * 8 + w) * 16 + lr;    // 0..895 -> m = col/128, c = col%128
      int m = col >> 7, c = col & 127;
      #pragma unroll
      for (int r = 0; r < 8; ++r) {
        int row = hi * 8 + r;
        float v = (acc[t][r] + b3[col]) * sScale[row];
        sM0[row * OUT3 + c * 7 + m] = f32_to_bf16u(v);
      }
    }
  }
  __syncthreads();

  // ---- einsum [49x7]@[7x128] per edge via WMMA (K=7 zero-padded to 32) + atomic scatter
  {
    const __bf16 zb = us2bf(0);
    for (int le = w * 2; le < w * 2 + 2; ++le) {
      int e = e0 + le;                            // wave-uniform
      int tgt = eidx[E_EDGES + e];
      float* outb = out + (size_t)tgt * (LFULL * SPH);
      const float* wigE = wig + (size_t)e * (LFULL * M_ALL);
      __builtin_prefetch(wigE, 0, 1);             // global_prefetch of wigner block
      const unsigned short* m0e = sM0 + le * OUT3;
      #pragma unroll 1
      for (int mt = 0; mt < 4; ++mt) {
        v16bf a;
        #pragma unroll
        for (int i = 0; i < 16; ++i) a[i] = zb;
        int row = mt * 16 + lr;
        if (hi == 0 && row < LFULL) {             // only lanes 0-15 carry K=0..6
          const float* wr = wigE + (size_t)row * M_ALL;
          #pragma unroll
          for (int m = 0; m < M0_DIM; ++m) a[m] = us2bf(f32_to_bf16u(wr[m]));
        }
        #pragma unroll 1
        for (int nt = 0; nt < 8; ++nt) {
          v16bf b;
          #pragma unroll
          for (int i = 0; i < 16; ++i) b[i] = zb;
          if (hi == 0) {
            const unsigned short* mc = m0e + (nt * 16 + lr) * 7;
            #pragma unroll
            for (int m = 0; m < M0_DIM; ++m) b[m] = us2bf(mc[m]);
          }
          v8f c = {0.f, 0.f, 0.f, 0.f, 0.f, 0.f, 0.f, 0.f};
          c = __builtin_amdgcn_wmma_f32_16x16x32_bf16(false, a, false, b, (short)0, c, false, false);
          int coln = nt * 16 + lr;
          #pragma unroll
          for (int r = 0; r < 8; ++r) {
            int gr = mt * 16 + hi * 8 + r;
            if (gr < LFULL) atomicAdd(outb + (size_t)gr * SPH + coln, c[r]);
          }
        }
      }
    }
  }
}

extern "C" void kernel_launch(void* const* d_in, const int* in_sizes, int n_in,
                              void* d_out, int out_size, void* d_ws, size_t ws_size,
                              hipStream_t stream) {
  const float* x     = (const float*)d_in[0];
  const float* ede   = (const float*)d_in[1];
  const float* semb  = (const float*)d_in[2];
  const float* temb  = (const float*)d_in[3];
  const float* edist = (const float*)d_in[4];
  const int*   eidx  = (const int*)  d_in[5];
  const float* wig   = (const float*)d_in[6];
  const float* W1    = (const float*)d_in[7];
  const float* b1    = (const float*)d_in[8];
  const float* g1    = (const float*)d_in[9];
  const float* bt1   = (const float*)d_in[10];
  const float* W2    = (const float*)d_in[11];
  const float* b2    = (const float*)d_in[12];
  const float* g2    = (const float*)d_in[13];
  const float* bt2   = (const float*)d_in[14];
  const float* W3    = (const float*)d_in[15];
  const float* b3    = (const float*)d_in[16];
  float* out = (float*)d_out;

  unsigned short* W1T = (unsigned short*)d_ws;
  unsigned short* W2T = W1T + (size_t)HID * D_IN;
  unsigned short* W3T = W2T + (size_t)HID * HID;

  convert_weights_kernel<<<240, 256, 0, stream>>>(W1, W2, W3, W1T, W2T, W3T);

  int n4 = NNODES * LFULL * SPH / 4;
  copy_x_kernel<<<(n4 + 255) / 256, 256, 0, stream>>>((const float4*)x, (float4*)out, n4);

  fused_edge_kernel<<<(E_EDGES + 15) / 16, 256, 0, stream>>>(
      ede, semb, temb, edist, eidx, wig,
      W1T, b1, g1, bt1, W2T, b2, g2, bt2, W3T, b3, out);
}
